// SynergyGT_26225070309508
// MI455X (gfx1250) — compile-verified, matrix operand order checked by reading, not run on hardware
//
#include <hip/hip_runtime.h>
#include <hip/hip_bf16.h>
#include <math.h>

// ---------------- problem constants ----------------
constexpr int DD = 256, H = 8, LAYERS = 4, E = 4;
constexpr int MAX_SPD = 6, MAX_LD = 10, NCLS = 8;
constexpr int B = 16, N = 511, T = N + 1;       // T = 512
constexpr int DH = DD / H;                       // 32
constexpr int FF = 4 * DD;                       // 1024
constexpr int BT = B * T;                        // 8192
constexpr float INV_SCALE = 0.17677669529663687f; // 1/sqrt(32)
constexpr float EPSF = 1e-5f;
constexpr float NEG_BIG = -3.0e38f;

// ---------------- vector types ----------------
typedef __attribute__((ext_vector_type(16))) _Float16 v16h;
typedef __attribute__((ext_vector_type(8)))  _Float16 v8h;
typedef __attribute__((ext_vector_type(8)))  float    v8f;

// Load a 16x32 f16 tile in the CDNA5 WMMA A-operand layout.
// Lane L (L = lo + 16*hi): row = lo, halves 0..7 -> K = 8*hi..8*hi+7,
// halves 8..15 -> K = 16+8*hi .. 16+8*hi+7.  Two 16-byte vector loads.
__device__ __forceinline__ v16h load_tileA(const _Float16* base, int stride) {
  int lane = threadIdx.x & 31;
  int lo = lane & 15, hi = lane >> 4;
  const _Float16* p = base + (long)lo * stride + 8 * hi;
  v8h a = *(const v8h*)(p);
  v8h b = *(const v8h*)(p + 16);
  return __builtin_shufflevector(a, b, 0,1,2,3,4,5,6,7,8,9,10,11,12,13,14,15);
}

__device__ __forceinline__ v8f wmma_f16(v16h a, v16h b, v8f c) {
  return __builtin_amdgcn_wmma_f32_16x16x32_f16(false, a, false, b, (short)0, c,
                                                false, false);
}

// ---------------- weight transpose fp32 -> f16 ----------------
// W is K x Nn row-major; Wt becomes Nn x K row-major (B-operand via A-layout).
__global__ void transpose_f16_kernel(const float* __restrict__ W,
                                     _Float16* __restrict__ Wt, int K, int Nn) {
  long idx = (long)blockIdx.x * 256 + threadIdx.x;
  long total = (long)K * Nn;
  if (idx >= total) return;
  int n = (int)(idx % Nn);
  long k = idx / Nn;
  Wt[(long)n * K + k] = (_Float16)W[idx];
}

// ---------------- embedding assembly ----------------
__global__ void embed_kernel(const int* __restrict__ node_ids,
                             const int* __restrict__ dtu, const int* __restrict__ dtv,
                             const int* __restrict__ indeg, const int* __restrict__ outdeg,
                             const int* __restrict__ life, const int* __restrict__ npert,
                             const int* __restrict__ ppert,
                             const float* __restrict__ node_id_e,
                             const float* __restrict__ dist_uv_e,
                             const float* __restrict__ in_deg_e,
                             const float* __restrict__ out_deg_e,
                             const float* __restrict__ pert_e,
                             const float* __restrict__ lifespan_e,
                             const float* __restrict__ base_pair,
                             float* __restrict__ x, _Float16* __restrict__ xh) {
  int bt = blockIdx.x, d = threadIdx.x;
  int b = bt / T, t = bt % T;
  float v;
  if (t == 0) {
    int p0 = ppert[b * 2 + 0], p1 = ppert[b * 2 + 1];
    v = base_pair[d] + pert_e[p0 * DD + d] + pert_e[p1 * DD + d];
  } else {
    long i = (long)b * N + (t - 1);
    int nid = node_ids[i];
    int du = min(dtu[i], MAX_SPD), dv = min(dtv[i], MAX_SPD);
    int ldr = min(life[i], MAX_LD);
    int ld = (ldr == -1) ? (MAX_LD + 1) : ldr;
    v = node_id_e[(long)nid * DD + d]
      + 0.5f * (dist_uv_e[du * DD + d] + dist_uv_e[dv * DD + d])
      + 0.5f * (in_deg_e[indeg[i] * DD + d] + out_deg_e[outdeg[i] * DD + d])
      + pert_e[npert[i] * DD + d] + lifespan_e[ld * DD + d];
  }
  x[(long)bt * DD + d] = v;
  xh[(long)bt * DD + d] = (_Float16)v;
}

// ---------------- generic WMMA GEMM: out = act(A[M,K] @ Wt^T + bias) ----------
// A: f16 row-major MxK.  Wt: f16 NxK row-major (pre-transposed weight).
// Store address = (m/Tb)*bs + (m%Tb)*ts + n*ns  (lets us emit V transposed).
// act: 0 none, 1 exact GELU, 2 relu.  out_f16: store f16 else f32.
__global__ void wmma_gemm_kernel(const _Float16* __restrict__ A,
                                 const _Float16* __restrict__ Wt,
                                 const float* __restrict__ bias, void* out,
                                 int M, int Nn, int K, int Tb, long bs, long ts,
                                 long ns, int act, int out_f16) {
  int lane = threadIdx.x;
  int lo = lane & 15, hi = lane >> 4;
  int mt = blockIdx.x, nt = blockIdx.y;
  const _Float16* Abase = A + (long)mt * 16 * K;
  v8f c[4] = {};
  for (int k = 0; k < K; k += 32) {
    v16h a = load_tileA(Abase + k, K);
#pragma unroll
    for (int t = 0; t < 4; ++t) {
      v16h bb = load_tileA(Wt + (long)(nt * 64 + t * 16) * K + k, K);
      c[t] = wmma_f16(a, bb, c[t]);
    }
  }
#pragma unroll
  for (int t = 0; t < 4; ++t) {
#pragma unroll
    for (int r = 0; r < 8; ++r) {
      int m = mt * 16 + r + 8 * hi;          // C layout: row = r + 8*hi
      int n = nt * 64 + t * 16 + lo;         //           col = lane&15
      float v = c[t][r] + bias[n];
      if (act == 1) v = 0.5f * v * (1.0f + erff(v * 0.70710678118654752f));
      else if (act == 2) v = fmaxf(v, 0.0f);
      long addr = (long)(m / Tb) * bs + (long)(m % Tb) * ts + (long)n * ns;
      if (out_f16) ((_Float16*)out)[addr] = (_Float16)v;
      else         ((float*)out)[addr] = v;
    }
  }
}

// ---------------- fused edge-masked attention ----------------
// One wave per (b, h, 16-row q tile).  Pass 1: scores+bias+mask -> LDS, rowmax.
// Pass 2: unnormalized exp, 5-way edge-partitioned P@V via WMMA, divide by Z.
__global__ void attention_kernel(const _Float16* __restrict__ Qh,
                                 const _Float16* __restrict__ Kh,
                                 const _Float16* __restrict__ Vt,
                                 const int* __restrict__ pairwise_dist,
                                 const float* __restrict__ aete,
                                 const int* __restrict__ dtu,
                                 const int* __restrict__ dtv,
                                 const unsigned char* __restrict__ pm,
                                 const float* __restrict__ pd_bias,
                                 const float* __restrict__ duv_bias,
                                 _Float16* __restrict__ attn_out) {
  __shared__ float s_lds[16 * T];
  __shared__ float red[16], red2[16];
  int lane = threadIdx.x;
  int lo = lane & 15, hi = lane >> 4;
  int blk = blockIdx.x;
  int qt = blk % (T / 16), bh = blk / (T / 16);
  int h = bh % H, b = bh / H;
  int q0 = qt * 16;

  v16h aq = load_tileA(Qh + ((long)(b * T + q0) * DD + h * DH), DD);

  float m8[8];
#pragma unroll
  for (int r = 0; r < 8; ++r) m8[r] = NEG_BIG;

  // ---- pass 1: biased/masked scores ----
  for (int kt = 0; kt < T / 16; ++kt) {
    v16h bk = load_tileA(Kh + ((long)(b * T + kt * 16) * DD + h * DH), DD);
    v8f c = {};
    c = wmma_f16(aq, bk, c);
    int j = kt * 16 + lo;                       // this lane's key column
    bool masked = (j >= 1) && (pm[(long)b * N + (j - 1)] != 0);
    float clsb = 0.0f;
    if (j >= 1) {
      int du = min(dtu[(long)b * N + (j - 1)], MAX_SPD);
      int dv = min(dtv[(long)b * N + (j - 1)], MAX_SPD);
      clsb = 0.5f * (duv_bias[du * H + h] + duv_bias[dv * H + h]);
    }
#pragma unroll
    for (int r = 0; r < 8; ++r) {
      int i = q0 + r + 8 * hi;                  // query row of this accum slot
      float sv;
      if (masked) {
        sv = NEG_BIG;
      } else {
        int pdv = (i == 0 || j == 0) ? (MAX_SPD + 1)
                  : pairwise_dist[((long)b * N + (i - 1)) * N + (j - 1)];
        float bias = pd_bias[pdv * H + h];
        if (i == 0 && j >= 1) bias += clsb;
        sv = c[r] * INV_SCALE + bias;
      }
      m8[r] = fmaxf(m8[r], sv);
      s_lds[(r + 8 * hi) * T + j] = sv;
    }
  }
  // row-max reduce across the 16 lanes holding each row
#pragma unroll
  for (int r = 0; r < 8; ++r) {
    float mv = m8[r];
    mv = fmaxf(mv, __shfl_xor(mv, 1));
    mv = fmaxf(mv, __shfl_xor(mv, 2));
    mv = fmaxf(mv, __shfl_xor(mv, 4));
    mv = fmaxf(mv, __shfl_xor(mv, 8));
    if (lo == 0) red[r + 8 * hi] = mv;
  }
  __syncthreads();

  // ---- pass 2: exp + edge-partitioned P@V ----
  v8f c0 = {}, c1 = {};
  float zacc = 0.0f;
  float mrow = red[lo];                         // A-operand row of this lane
  for (int kt2 = 0; kt2 < T / 32; ++kt2) {
    float ev[16];
    int av[16];
#pragma unroll
    for (int e2 = 0; e2 < 16; ++e2) {
      int k = kt2 * 32 + ((e2 < 8) ? (8 * hi + e2) : (16 + 8 * hi + (e2 - 8)));
      float ex = __expf(s_lds[lo * T + k] - mrow);
      ev[e2] = ex;
      zacc += ex;
      int i = q0 + lo;
      av[e2] = (i == 0 || k == 0)
                   ? E
                   : (int)rintf(aete[((long)b * N + (i - 1)) * N + (k - 1)]);
    }
    for (int e = 0; e <= E; ++e) {
      v16h pa;
#pragma unroll
      for (int e2 = 0; e2 < 16; ++e2)
        pa[e2] = (av[e2] == e) ? (_Float16)ev[e2] : (_Float16)0.0f;
      const _Float16* vb =
          Vt + (((long)e * B + b) * DD + h * DH) * T + kt2 * 32;
      v16h b0 = load_tileA(vb, T);              // dims 0..15  (rows of V^T)
      v16h b1 = load_tileA(vb + 16L * T, T);    // dims 16..31
      c0 = wmma_f16(pa, b0, c0);
      c1 = wmma_f16(pa, b1, c1);
    }
  }
  zacc += __shfl_xor(zacc, 16);                 // combine the two key halves
  if (hi == 0) red2[lo] = zacc;
  __syncthreads();
#pragma unroll
  for (int r = 0; r < 8; ++r) {
    float inv = 1.0f / red2[r + 8 * hi];
    long base = ((long)(b * T) + q0 + r + 8 * hi) * DD + h * DH + lo;
    attn_out[base]      = (_Float16)(c0[r] * inv);
    attn_out[base + 16] = (_Float16)(c1[r] * inv);
  }
}

// ---------------- residual + LayerNorm (f32 + f16 copies out) ----------------
__global__ void ln_residual_kernel(const float* __restrict__ x,
                                   const float* __restrict__ y,
                                   const float* __restrict__ g,
                                   const float* __restrict__ be,
                                   float* __restrict__ xo,
                                   _Float16* __restrict__ xoh) {
  int row = blockIdx.x, tid = threadIdx.x;      // 256 threads, D=256
  __shared__ float warpsum[8];
  __shared__ float stats[2];
  float v = x[(long)row * DD + tid] + y[(long)row * DD + tid];
  float s = v;
  for (int m = 1; m < 32; m <<= 1) s += __shfl_xor(s, m);
  if ((tid & 31) == 0) warpsum[tid >> 5] = s;
  __syncthreads();
  if (tid < 8) {
    float t2 = warpsum[tid];
    for (int m = 1; m < 8; m <<= 1) t2 += __shfl_xor(t2, m);
    if (tid == 0) stats[0] = t2 / DD;
  }
  __syncthreads();
  float dv = v - stats[0];
  float q = dv * dv;
  for (int m = 1; m < 32; m <<= 1) q += __shfl_xor(q, m);
  if ((tid & 31) == 0) warpsum[tid >> 5] = q;
  __syncthreads();
  if (tid < 8) {
    float t2 = warpsum[tid];
    for (int m = 1; m < 8; m <<= 1) t2 += __shfl_xor(t2, m);
    if (tid == 0) stats[1] = t2 / DD;
  }
  __syncthreads();
  float o = dv * rsqrtf(stats[1] + EPSF) * g[tid] + be[tid];
  xo[(long)row * DD + tid] = o;
  xoh[(long)row * DD + tid] = (_Float16)o;
}

// ---------------- classification head (tiny, plain VALU) ----------------
__global__ void head_kernel(const float* __restrict__ x,
                            const float* __restrict__ Wc1,
                            const float* __restrict__ bc1,
                            const float* __restrict__ Wc2,
                            const float* __restrict__ bc2,
                            float* __restrict__ out) {
  __shared__ float hbuf[2 * DD];
  int b = blockIdx.x, n = threadIdx.x;          // blockDim = 512
  const float* rep = x + (long)b * T * DD;      // row t=0
  float acc = bc1[n];
  for (int d = 0; d < DD; ++d) acc += rep[d] * Wc1[d * 2 * DD + n];
  hbuf[n] = fmaxf(acc, 0.0f);
  __syncthreads();
  if (n < NCLS) {
    float o = bc2[n];
    for (int k = 0; k < 2 * DD; ++k) o += hbuf[k] * Wc2[k * NCLS + n];
    out[b * NCLS + n] = o;
  }
}

// ---------------- input indexing (jax pytree flatten: dict keys sorted) ------
enum {
  IN_NODE_IDS = 0, IN_DTU, IN_DTV, IN_INDEG, IN_OUTDEG, IN_LIFE, IN_NPERT,
  IN_PPERT, IN_PD, IN_AETE, IN_PM,
  P_WC1, P_WC2, P_WV, P_BASEPAIR, P_BC1, P_BC2, P_BV, P_DISTUV, P_DUVBIAS,
  P_INDEG,
  P_LAYER0,                       // 56 leaves: 14 per layer x 4 layers
  P_LIFESPAN = P_LAYER0 + 56, P_NODEID, P_OUTDEG, P_PDBIAS, P_PERT
};
// per-layer sorted keys: W1 W2 Wk Wo Wq b1 b2 be1 be2 bk bo bq g1 g2
enum { L_W1 = 0, L_W2, L_WK, L_WO, L_WQ, L_B1, L_B2, L_BE1, L_BE2, L_BK,
       L_BO, L_BQ, L_G1, L_G2 };

static inline void launch_transpose(const float* W, _Float16* Wt, int K, int Nn,
                                    hipStream_t s) {
  long total = (long)K * Nn;
  transpose_f16_kernel<<<dim3((unsigned)((total + 255) / 256)), 256, 0, s>>>(
      W, Wt, K, Nn);
}

static inline void launch_gemm(const _Float16* A, const _Float16* Wt,
                               const float* bias, void* out, int M, int Nn,
                               int K, int Tb, long bs, long ts, long ns,
                               int act, int out_f16, hipStream_t s) {
  wmma_gemm_kernel<<<dim3(M / 16, Nn / 64), 32, 0, s>>>(
      A, Wt, bias, out, M, Nn, K, Tb, bs, ts, ns, act, out_f16);
}

extern "C" void kernel_launch(void* const* d_in, const int* in_sizes, int n_in,
                              void* d_out, int out_size, void* d_ws,
                              size_t ws_size, hipStream_t stream) {
  (void)in_sizes; (void)n_in; (void)out_size; (void)ws_size;
  // -------- workspace carve (all 256B aligned) --------
  char* p = (char*)d_ws;
  auto carve = [&](size_t bytes) {
    char* r = p;
    p += (bytes + 255) & ~(size_t)255;
    return (void*)r;
  };
  float*    x    = (float*)   carve((size_t)BT * DD * 4);
  float*    y    = (float*)   carve((size_t)BT * DD * 4);
  _Float16* xh   = (_Float16*)carve((size_t)BT * DD * 2);
  _Float16* Qh   = (_Float16*)carve((size_t)BT * DD * 2);
  _Float16* Kh   = (_Float16*)carve((size_t)BT * DD * 2);
  _Float16* Vt   = (_Float16*)carve((size_t)(E + 1) * BT * DD * 2);
  _Float16* attn = (_Float16*)carve((size_t)BT * DD * 2);
  _Float16* ffh  = (_Float16*)carve((size_t)BT * FF * 2);
  _Float16* wt   = (_Float16*)carve((size_t)DD * FF * 2);  // reused per weight

  const int* node_ids = (const int*)d_in[IN_NODE_IDS];
  const int* dtu      = (const int*)d_in[IN_DTU];
  const int* dtv      = (const int*)d_in[IN_DTV];
  const int* indeg    = (const int*)d_in[IN_INDEG];
  const int* outdeg   = (const int*)d_in[IN_OUTDEG];
  const int* life     = (const int*)d_in[IN_LIFE];
  const int* npert    = (const int*)d_in[IN_NPERT];
  const int* ppert    = (const int*)d_in[IN_PPERT];
  const int* pd       = (const int*)d_in[IN_PD];
  const float* aete   = (const float*)d_in[IN_AETE];
  const unsigned char* pm = (const unsigned char*)d_in[IN_PM];

  // -------- embeddings --------
  embed_kernel<<<dim3(BT), DD, 0, stream>>>(
      node_ids, dtu, dtv, indeg, outdeg, life, npert, ppert,
      (const float*)d_in[P_NODEID], (const float*)d_in[P_DISTUV],
      (const float*)d_in[P_INDEG], (const float*)d_in[P_OUTDEG],
      (const float*)d_in[P_PERT], (const float*)d_in[P_LIFESPAN],
      (const float*)d_in[P_BASEPAIR], x, xh);

  const float* Wv = (const float*)d_in[P_WV];
  const float* bv = (const float*)d_in[P_BV];

  for (int l = 0; l < LAYERS; ++l) {
    const int base = P_LAYER0 + l * 14;
    const float* Wq = (const float*)d_in[base + L_WQ];
    const float* Wk = (const float*)d_in[base + L_WK];
    const float* Wo = (const float*)d_in[base + L_WO];
    const float* W1 = (const float*)d_in[base + L_W1];
    const float* W2 = (const float*)d_in[base + L_W2];
    const float* bq = (const float*)d_in[base + L_BQ];
    const float* bk = (const float*)d_in[base + L_BK];
    const float* bo = (const float*)d_in[base + L_BO];
    const float* b1 = (const float*)d_in[base + L_B1];
    const float* b2 = (const float*)d_in[base + L_B2];
    const float* g1 = (const float*)d_in[base + L_G1];
    const float* be1 = (const float*)d_in[base + L_BE1];
    const float* g2 = (const float*)d_in[base + L_G2];
    const float* be2 = (const float*)d_in[base + L_BE2];

    // Q, K projections (f16, natural [BT, D] layout)
    launch_transpose(Wq, wt, DD, DD, stream);
    launch_gemm(xh, wt, bq, Qh, BT, DD, DD, BT, 0, DD, 1, 0, 1, stream);
    launch_transpose(Wk, wt, DD, DD, stream);
    launch_gemm(xh, wt, bk, Kh, BT, DD, DD, BT, 0, DD, 1, 0, 1, stream);

    // V_e projections, stored TRANSPOSED: Vt[e][b][n][t]
    for (int e = 0; e <= E; ++e) {
      launch_transpose(Wv + (long)e * DD * DD, wt, DD, DD, stream);
      launch_gemm(xh, wt, bv + (long)e * DD, Vt + (long)e * B * DD * T,
                  BT, DD, DD, T, (long)DD * T, 1, T, 0, 1, stream);
    }

    // fused biased/masked attention with edge-partitioned V
    attention_kernel<<<dim3(B * H * (T / 16)), 32, 0, stream>>>(
        Qh, Kh, Vt, pd, aete, dtu, dtv, pm,
        (const float*)d_in[P_PDBIAS], (const float*)d_in[P_DUVBIAS], attn);

    // output projection -> f32, then residual + LN (also emits f16 x)
    launch_transpose(Wo, wt, DD, DD, stream);
    launch_gemm(attn, wt, bo, y, BT, DD, DD, BT, 0, DD, 1, 0, 0, stream);
    ln_residual_kernel<<<dim3(BT), DD, 0, stream>>>(x, y, g1, be1, x, xh);

    // FFN: GELU(x@W1+b1) @ W2 + b2, residual + LN
    launch_transpose(W1, wt, DD, FF, stream);
    launch_gemm(xh, wt, b1, ffh, BT, FF, DD, BT, 0, FF, 1, 1, 1, stream);
    launch_transpose(W2, wt, FF, DD, stream);
    launch_gemm(ffh, wt, b2, y, BT, DD, FF, BT, 0, DD, 1, 0, 0, stream);
    ln_residual_kernel<<<dim3(BT), DD, 0, stream>>>(x, y, g2, be2, x, xh);
  }

  // -------- classification head --------
  head_kernel<<<dim3(B), 2 * DD, 0, stream>>>(
      x, (const float*)d_in[P_WC1], (const float*)d_in[P_BC1],
      (const float*)d_in[P_WC2], (const float*)d_in[P_BC2], (float*)d_out);
}
